// ResidualVectorQuantizer_76270029242595
// MI455X (gfx1250) — matrix operand order repaired and checked by reading, not compiled, not run on hardware
//
#include <hip/hip_runtime.h>

// ---------------- problem constants (match reference) ----------------
#define B_ROWS 16384
#define DIMS   512
#define KCW    8192
#define NLEV   3
#define BETA_C   0.25f
#define USAGE_RG 0.001f
#define EPS_C    1e-5f
#define LOG_K  9.0109131f     // ln(8192)
#define LOG_B  9.7040605f     // ln(16384)

// GEMM tiling
#define BLK_ROWS 32     // rows per workgroup
#define WG_COLS  256    // 8 waves * 32 cols
#define KCHUNKS  16     // DIMS / 32
#define NITER    32     // KCW / WG_COLS

typedef __attribute__((ext_vector_type(16))) __bf16 v16bf;
typedef __attribute__((ext_vector_type(8)))  float  v8f;

union Frag { v16bf v; uint4 u[2]; };

static __device__ inline unsigned short f32_to_bf16(float f) {
  unsigned u = __float_as_uint(f);
  unsigned r = 0x7FFFu + ((u >> 16) & 1u);   // round to nearest even
  return (unsigned short)((u + r) >> 16);
}
static __device__ inline float bf16_to_f32(unsigned short h) {
  return __uint_as_float(((unsigned)h) << 16);
}
static __device__ inline v8f zero8() {
  v8f z;
#pragma unroll
  for (int i = 0; i < 8; ++i) z[i] = 0.0f;
  return z;
}

// ---------------- small kernels ----------------

__global__ void init_scalars_kernel(float* scal) {
  if (threadIdx.x < 4) scal[threadIdx.x] = 0.0f;
}

// one block per codeword row: split fp32 -> bf16 hi/lo planes, compute ||e||^2
__global__ void split_cb_kernel(const float* __restrict__ cb,
                                unsigned short* __restrict__ hi,
                                unsigned short* __restrict__ lo,
                                float* __restrict__ e2) {
  __shared__ float red[256];
  const int row = blockIdx.x;
  const int tid = threadIdx.x;
  float acc = 0.0f;
#pragma unroll
  for (int i = tid; i < DIMS; i += 256) {
    float f = cb[(size_t)row * DIMS + i];
    unsigned short h = f32_to_bf16(f);
    float fh = bf16_to_f32(h);
    unsigned short l = f32_to_bf16(f - fh);
    hi[(size_t)row * DIMS + i] = h;
    lo[(size_t)row * DIMS + i] = l;
    acc += f * f;
  }
  red[tid] = acc;
  __syncthreads();
  for (int s = 128; s > 0; s >>= 1) {
    if (tid < s) red[tid] += red[tid + s];
    __syncthreads();
  }
  if (tid == 0) e2[row] = red[0];
}

// elementwise residual split; first 32 blocks also zero avg_probs
__global__ void split_res_kernel(const float* __restrict__ r,
                                 unsigned short* __restrict__ hi,
                                 unsigned short* __restrict__ lo,
                                 float* __restrict__ avgp) {
  const size_t i = (size_t)blockIdx.x * 256 + threadIdx.x;
  float f = r[i];
  unsigned short h = f32_to_bf16(f);
  float fh = bf16_to_f32(h);
  unsigned short l = f32_to_bf16(f - fh);
  hi[i] = h;
  lo[i] = l;
  if (blockIdx.x < (KCW / 256)) avgp[blockIdx.x * 256 + threadIdx.x] = 0.0f;
}

// ---------------- fused distance GEMM ----------------
// Works on d' = ||e||^2 - 2*dot (per-row constant ||r||^2 dropped: argmin,
// softmax denominator and exp(m-d) are invariant to it).
// PASS 1: per-row min / argmin / online softmax denominator
// PASS 2: recompute dots, accumulate avg_probs[k] += exp(mb - d'),
//         mb = m - log(s) - log(B)
template <int PASS>
__global__ __launch_bounds__(256)
void rvq_gemm(const unsigned short* __restrict__ r_hi,
              const unsigned short* __restrict__ r_lo,
              const unsigned short* __restrict__ cb_hi,
              const unsigned short* __restrict__ cb_lo,
              const float* __restrict__ e2,
              float* __restrict__ m_arr,
              float* __restrict__ s_arr,
              unsigned* __restrict__ idx_arr,
              float* __restrict__ avg_probs) {
  // A staged pre-swizzled into WMMA fragment order:
  // chunk index = plane*2048 + kc*128 + rt*64 + load*32 + lane, 16B per chunk
  __shared__ __align__(16) unsigned char smem[65536];
  uint4* sA = (uint4*)smem;

  const int tid  = threadIdx.x;
  const int wave = tid >> 5;
  const int lane = tid & 31;
  const int half = lane >> 4;
  const int lq   = lane & 15;
  const int row0 = blockIdx.x * BLK_ROWS;

  // ---- stage A (both planes) into LDS in fragment layout ----
  for (int c = tid; c < 4096; c += 256) {
    int plane = c >> 11;
    int kc    = (c >> 7) & 15;
    int rt    = (c >> 6) & 1;
    int ld    = (c >> 5) & 1;
    int ln    = c & 31;
    int rrow  = rt * 16 + (ln & 15);
    int ebase = kc * 32 + ((ln >> 4) ? 8 : 0) + ld * 16;
    const unsigned short* src =
        (plane ? r_lo : r_hi) + (size_t)(row0 + rrow) * DIMS + ebase;
    sA[c] = *(const uint4*)src;
  }

  float m_st[16], s_st[16];
  unsigned i_st[16];
  float mb[16];
  if (PASS == 1) {
#pragma unroll
    for (int st = 0; st < 16; ++st) { m_st[st] = 3.0e38f; s_st[st] = 0.0f; i_st[st] = 0u; }
  } else {
#pragma unroll
    for (int rt = 0; rt < 2; ++rt)
#pragma unroll
      for (int v = 0; v < 8; ++v) {
        int row = row0 + rt * 16 + half * 8 + v;
        mb[rt * 8 + v] = m_arr[row] - __logf(s_arr[row]) - LOG_B;
      }
  }

  __syncthreads();

  for (int it = 0; it < NITER; ++it) {
    // LDS fence each iteration: keeps the A-fragment ds_loads inside the
    // column loop (prevents whole-A promotion to VGPRs + scratch spills).
    __syncthreads();

    const int colbase = it * WG_COLS + wave * 32;

    v8f acc[2][2];
#pragma unroll
    for (int rt = 0; rt < 2; ++rt)
#pragma unroll
      for (int ct = 0; ct < 2; ++ct) acc[rt][ct] = zero8();

#pragma unroll 4
    for (int kc = 0; kc < KCHUNKS; ++kc) {
      Frag ah[2], al[2];
#pragma unroll
      for (int rt = 0; rt < 2; ++rt) {
        int b0 = kc * 128 + rt * 64 + lane;       // plane 0 (hi)
        ah[rt].u[0] = sA[b0];
        ah[rt].u[1] = sA[b0 + 32];
        int b1 = b0 + 2048;                       // plane 1 (lo)
        al[rt].u[0] = sA[b1];
        al[rt].u[1] = sA[b1 + 32];
      }
#pragma unroll
      for (int ct = 0; ct < 2; ++ct) {
        Frag bh, bl;
        int col = colbase + ct * 16 + lq;
        size_t off = (size_t)col * DIMS + kc * 32 + half * 8;
        bh.u[0] = *(const uint4*)(cb_hi + off);
        bh.u[1] = *(const uint4*)(cb_hi + off + 16);
        bl.u[0] = *(const uint4*)(cb_lo + off);
        bl.u[1] = *(const uint4*)(cb_lo + off + 16);
#pragma unroll
        for (int rt = 0; rt < 2; ++rt) {
          acc[rt][ct] = __builtin_amdgcn_wmma_f32_16x16x32_bf16(
              false, ah[rt].v, false, bh.v, (short)0, acc[rt][ct], false, false);
          acc[rt][ct] = __builtin_amdgcn_wmma_f32_16x16x32_bf16(
              false, ah[rt].v, false, bl.v, (short)0, acc[rt][ct], false, false);
          acc[rt][ct] = __builtin_amdgcn_wmma_f32_16x16x32_bf16(
              false, al[rt].v, false, bh.v, (short)0, acc[rt][ct], false, false);
        }
      }
    }

    // ---- epilogue for this 32x(2x16) tile ----
    float e2t[2];
#pragma unroll
    for (int ct = 0; ct < 2; ++ct) e2t[ct] = e2[colbase + ct * 16 + lq];

    if (PASS == 1) {
#pragma unroll
      for (int rt = 0; rt < 2; ++rt)
#pragma unroll
        for (int ct = 0; ct < 2; ++ct) {
          int col = colbase + ct * 16 + lq;
#pragma unroll
          for (int v = 0; v < 8; ++v) {
            float d = e2t[ct] - 2.0f * acc[rt][ct][v];
            int st = rt * 8 + v;
            float e = __expf(-fabsf(d - m_st[st]));
            if (d < m_st[st]) {
              s_st[st] = s_st[st] * e + 1.0f;
              m_st[st] = d;
              i_st[st] = (unsigned)col;
            } else {
              s_st[st] += e;
            }
          }
        }
    } else {
      float colsum[2] = {0.0f, 0.0f};
#pragma unroll
      for (int rt = 0; rt < 2; ++rt)
#pragma unroll
        for (int ct = 0; ct < 2; ++ct)
#pragma unroll
          for (int v = 0; v < 8; ++v) {
            float d = e2t[ct] - 2.0f * acc[rt][ct][v];
            colsum[ct] += __expf(mb[rt * 8 + v] - d);
          }
#pragma unroll
      for (int ct = 0; ct < 2; ++ct) {
        float o = __shfl_xor(colsum[ct], 16, 32);  // add lane-pair (other row half)
        colsum[ct] += o;
        if (half == 0)
          atomicAdd(&avg_probs[colbase + ct * 16 + lq], colsum[ct]);
      }
    }
  }

  if (PASS == 1) {
    // ---- merge per-lane online states across the workgroup (reuse smem) ----
    __syncthreads();
    float*    Lm = (float*)smem;            // [32][128]
    float*    Ls = Lm + 32 * 128;
    unsigned* Li = (unsigned*)(Ls + 32 * 128);
    const int cid = wave * 16 + lq;
#pragma unroll
    for (int st = 0; st < 16; ++st) {
      int rl = (st >> 3) * 16 + half * 8 + (st & 7);
      Lm[rl * 128 + cid] = m_st[st];
      Ls[rl * 128 + cid] = s_st[st];
      Li[rl * 128 + cid] = i_st[st];
    }
    __syncthreads();
    if (tid < BLK_ROWS) {
      float M = 3.0e38f, S = 0.0f;
      unsigned I = 0u;
      for (int c = 0; c < 128; ++c) {
        float mc = Lm[tid * 128 + c];
        float sc = Ls[tid * 128 + c];
        unsigned ic = Li[tid * 128 + c];
        float e = __expf(-fabsf(mc - M));
        if (mc < M) { S = S * e + sc; M = mc; I = ic; }
        else        { S += sc * e; }
      }
      m_arr[row0 + tid] = M;
      s_arr[row0 + tid] = S;
      idx_arr[row0 + tid] = I;
    }
  }
}

// gather codeword, update residual, commit-loss partial, emit code
__global__ void finalize_kernel(float* __restrict__ r,
                                const float* __restrict__ cb_level,
                                const unsigned* __restrict__ idx_arr,
                                float* __restrict__ commit_acc,
                                float* __restrict__ out_codes,
                                int level) {
  __shared__ float red[256];
  const int row = blockIdx.x;
  const int tid = threadIdx.x;
  const unsigned idx = idx_arr[row];
  const float* q = cb_level + (size_t)idx * DIMS;
  float acc = 0.0f;
#pragma unroll
  for (int i = tid; i < DIMS; i += 256) {
    float nr = r[(size_t)row * DIMS + i] - q[i];
    r[(size_t)row * DIMS + i] = nr;
    acc += nr * nr;
  }
  red[tid] = acc;
  __syncthreads();
  for (int s = 128; s > 0; s >>= 1) {
    if (tid < s) red[tid] += red[tid + s];
    __syncthreads();
  }
  if (tid == 0) {
    atomicAdd(commit_acc, red[0]);
    out_codes[(size_t)row * NLEV + level] = (float)idx;
  }
}

// entropy of avg_probs -> usage loss accumulator
__global__ void usage_kernel(const float* __restrict__ avgp, float* usage_acc) {
  __shared__ float red[256];
  const int tid = threadIdx.x;
  float acc = 0.0f;
  for (int k = tid; k < KCW; k += 256) {
    float p = avgp[k];
    p = (p < EPS_C) ? EPS_C : p;
    acc += p * logf(p);            // = -entropy contribution
  }
  red[tid] = acc;
  __syncthreads();
  for (int s = 128; s > 0; s >>= 1) {
    if (tid < s) red[tid] += red[tid + s];
    __syncthreads();
  }
  if (tid == 0) *usage_acc += USAGE_RG * (LOG_K + red[0]);
}

// quantized_final = x - residual_final ; write scalar losses
__global__ void final_out_kernel(const float* __restrict__ x,
                                 const float* __restrict__ r,
                                 float* __restrict__ out,
                                 const float* __restrict__ scal) {
  const size_t i = (size_t)blockIdx.x * 256 + threadIdx.x;
  out[i] = x[i] - r[i];
  if (i == 0) {
    const size_t base = (size_t)B_ROWS * DIMS + (size_t)B_ROWS * NLEV;
    float commit = BETA_C * (scal[0] + scal[1] + scal[2]) /
                   ((float)B_ROWS * (float)DIMS) / (float)NLEV;
    out[base]     = commit;
    out[base + 1] = scal[3] / (float)NLEV;
  }
}

// ---------------- host launcher ----------------
extern "C" void kernel_launch(void* const* d_in, const int* in_sizes, int n_in,
                              void* d_out, int out_size, void* d_ws, size_t ws_size,
                              hipStream_t stream) {
  const float* x  = (const float*)d_in[0];
  const float* cb = (const float*)d_in[1];
  float* out = (float*)d_out;
  char*  ws  = (char*)d_ws;

  float*          r_f32  = (float*)(ws + 0ULL);                 // 32 MB
  unsigned short* r_hi   = (unsigned short*)(ws + 33554432ULL); // 16 MB
  unsigned short* r_lo   = (unsigned short*)(ws + 50331648ULL); // 16 MB
  unsigned short* cb_hi  = (unsigned short*)(ws + 67108864ULL); // 24 MB
  unsigned short* cb_lo  = (unsigned short*)(ws + 92274688ULL); // 24 MB
  float*          e2     = (float*)(ws + 117440512ULL);
  float*          m_arr  = (float*)(ws + 117604352ULL);
  float*          s_arr  = (float*)(ws + 117669888ULL);
  unsigned*       idx_a  = (unsigned*)(ws + 117735424ULL);
  float*          avgp   = (float*)(ws + 117800960ULL);
  float*          scal   = (float*)(ws + 117833728ULL);

  hipMemcpyAsync(r_f32, x, (size_t)B_ROWS * DIMS * sizeof(float),
                 hipMemcpyDeviceToDevice, stream);
  init_scalars_kernel<<<1, 32, 0, stream>>>(scal);
  split_cb_kernel<<<NLEV * KCW, 256, 0, stream>>>(cb, cb_hi, cb_lo, e2);

  for (int lev = 0; lev < NLEV; ++lev) {
    split_res_kernel<<<(B_ROWS * DIMS) / 256, 256, 0, stream>>>(r_f32, r_hi, r_lo, avgp);
    const unsigned short* cbh = cb_hi + (size_t)lev * KCW * DIMS;
    const unsigned short* cbl = cb_lo + (size_t)lev * KCW * DIMS;
    const float* e2l = e2 + (size_t)lev * KCW;
    rvq_gemm<1><<<B_ROWS / BLK_ROWS, 256, 0, stream>>>(
        r_hi, r_lo, cbh, cbl, e2l, m_arr, s_arr, idx_a, avgp);
    rvq_gemm<2><<<B_ROWS / BLK_ROWS, 256, 0, stream>>>(
        r_hi, r_lo, cbh, cbl, e2l, m_arr, s_arr, idx_a, avgp);
    finalize_kernel<<<B_ROWS, 256, 0, stream>>>(
        r_f32, cb + (size_t)lev * KCW * DIMS, idx_a, scal + lev,
        out + (size_t)B_ROWS * DIMS, lev);
    usage_kernel<<<1, 256, 0, stream>>>(avgp, scal + 3);
  }

  final_out_kernel<<<(B_ROWS * DIMS) / 256, 256, 0, stream>>>(x, r_f32, out, scal);
}
